// CPUQuantizedLinear_64347200029419
// MI455X (gfx1250) — compile-verified, matrix operand order checked by reading, not run on hardware
//
#include <hip/hip_runtime.h>
#include <hip/hip_bf16.h>

// Problem dims (fixed by the reference)
#define M_TOT 8192      // B*S = 4*2048
#define N_TOT 11008     // OUT
#define K_TOT 4096      // IN
#define KPACK (K_TOT/2) // packed int4 elements per weight row

// Tiling: 128(M) x 256(N) block, 8 waves in 2x4, each wave owns 64x64
#define BM 128
#define BN 256
#define BK 64           // two wmma k-steps (16x16x32 bf16) per LDS stage
#define BKP (BK + 8)    // pad rows to 144B: fragment loads cover all 64 banks

#define GRID_MB (M_TOT / BM)   // 64 block rows
#define GRID_NB (N_TOT / BN)   // 43 block cols
#define SUPER_M 8              // L2 panel: 8 block-rows x all 43 block-cols

typedef __attribute__((ext_vector_type(16))) __bf16          v16bf;
typedef __attribute__((ext_vector_type(8)))  float           v8f;
typedef __attribute__((ext_vector_type(8)))  unsigned short  v8u;

union FragU {
    v8u   h[2];   // two 16-byte LDS loads
    v16bf bf;     // WMMA operand view
};

// Pack two f32 -> packed 2xbf16 with one v_perm_b32 (round-half-up via +0x8000)
__device__ __forceinline__ unsigned pack_bf16(float e, float o) {
    unsigned ue = __float_as_uint(e) + 0x8000u;
    unsigned uo = __float_as_uint(o) + 0x8000u;
    // result = [ue>>16] | [uo & 0xFFFF0000]
    return __builtin_amdgcn_perm(uo, ue, 0x07060302u);
}

// One packed byte (two nibbles) -> packed 2xbf16; v_cvt_f32_u32 is exact here
__device__ __forceinline__ unsigned pack_nib_bf16(int b) {
    float lo = (float)(b & 15);
    float hi = (float)((b >> 4) & 15);
    return __builtin_amdgcn_perm(__float_as_uint(hi), __float_as_uint(lo),
                                 0x07060302u);
}

__global__ __launch_bounds__(256)
void q4_bf16_wmma_gemm(const float* __restrict__ x,
                       const int*   __restrict__ wq,
                       const float* __restrict__ scales,
                       const int*   __restrict__ zps,
                       const float* __restrict__ bias,
                       float*       __restrict__ out)
{
    __shared__ unsigned short xs [BM][BKP];  // x tile, bf16 bits      (18 KB)
    __shared__ unsigned short wsh[BN][BKP];  // nibble tile, bf16 bits (36 KB)
    __shared__ float          rsum[BM];      // per-row sum of x over K

    const int tid  = threadIdx.x;
    const int lane = tid & 31;
    const int wave = tid >> 5;      // 0..7
    const int mw   = wave >> 2;     // 0..1 : wave row (64 rows)
    const int nw   = wave & 3;      // 0..3 : wave col (64 cols)
    const int half = lane >> 4;     // K-interleave half per WMMA layout
    const int l16  = lane & 15;

    // Supertile remap for L2 residency: walk 8 block-rows x 43 block-cols
    // per panel so x (134MB) stays L2-resident and weights stream 8x, not 64x.
    const int id    = (int)blockIdx.x;
    const int panel = id / (SUPER_M * GRID_NB);
    const int rem   = id % (SUPER_M * GRID_NB);
    const int bm    = (panel * SUPER_M + (rem % SUPER_M)) * BM;
    const int bn    = (rem / SUPER_M) * BN;

    // x loader: 2 threads per row, 32 K-elements each
    const int xrowl = tid >> 1;          // 0..127
    const int xk    = (tid & 1) * 32;    // 0 or 32
    // w loader: 1 thread per row, full BK (32 packed ints) each

    if (tid < BM) rsum[tid] = 0.0f;
    __syncthreads();

    v8f acc[4][4];
#pragma unroll
    for (int i = 0; i < 4; ++i)
#pragma unroll
        for (int j = 0; j < 4; ++j) acc[i][j] = (v8f)0.0f;

    const float* xp0 = x  + (size_t)(bm + xrowl) * K_TOT + xk;
    const int*   wp0 = wq + (size_t)(bn + tid) * KPACK;

    for (int kt = 0; kt < K_TOT / BK; ++kt) {
        const float* xp = xp0 + (size_t)kt * BK;
        const int*   wp = wp0 + kt * (BK / 2);

        if (kt + 1 < K_TOT / BK) {          // global_prefetch next stage
            __builtin_prefetch(xp + BK, 0, 0);
            __builtin_prefetch(wp + BK / 2, 0, 0);
        }

        // ---- stage x: fp32 -> bf16 (v_perm pack) -> LDS + row sums ----
        float psum = 0.0f;
#pragma unroll
        for (int q = 0; q < 4; ++q) {
            float4 t0 = *(const float4*)(xp + 8 * q);
            float4 t1 = *(const float4*)(xp + 8 * q + 4);
            psum += (t0.x + t0.y) + (t0.z + t0.w)
                  + ((t1.x + t1.y) + (t1.z + t1.w));
            uint4 c;
            c.x = pack_bf16(t0.x, t0.y);
            c.y = pack_bf16(t0.z, t0.w);
            c.z = pack_bf16(t1.x, t1.y);
            c.w = pack_bf16(t1.z, t1.w);
            *(uint4*)&xs[xrowl][xk + 8 * q] = c;
        }
        atomicAdd(&rsum[xrowl], psum);   // ds_add_f32

        // ---- stage w: unpack nibbles -> bf16 -> LDS ----
#pragma unroll
        for (int q = 0; q < 8; ++q) {
            int4 pk = *(const int4*)(wp + 4 * q);  // 4 bytes -> 8 k-values
            uint4 c;
            c.x = pack_nib_bf16(pk.x);
            c.y = pack_nib_bf16(pk.y);
            c.z = pack_nib_bf16(pk.z);
            c.w = pack_nib_bf16(pk.w);
            *(uint4*)&wsh[tid][8 * q] = c;
        }
        __syncthreads();

        // ---- compute: 2 k-steps x (4 m-tiles x 4 n-tiles) WMMA ----
#pragma unroll
        for (int ks = 0; ks < 2; ++ks) {
            FragU a[4], b[4];
#pragma unroll
            for (int i = 0; i < 4; ++i) {
                const unsigned short* p =
                    &xs[mw * 64 + i * 16 + l16][ks * 32 + half * 8];
                a[i].h[0] = *(const v8u*)p;
                a[i].h[1] = *(const v8u*)(p + 16);
            }
#pragma unroll
            for (int j = 0; j < 4; ++j) {
                const unsigned short* p =
                    &wsh[nw * 64 + j * 16 + l16][ks * 32 + half * 8];
                b[j].h[0] = *(const v8u*)p;
                b[j].h[1] = *(const v8u*)(p + 16);
            }
#pragma unroll
            for (int i = 0; i < 4; ++i)
#pragma unroll
                for (int j = 0; j < 4; ++j)
                    acc[i][j] = __builtin_amdgcn_wmma_f32_16x16x32_bf16(
                        false, a[i].bf, false, b[j].bf,
                        (short)0, acc[i][j], false, false);
        }
        __syncthreads();
    }

    // ---- epilogue: out = scale*(acc - zp*rowsum) + bias (streaming NT) ----
#pragma unroll
    for (int j = 0; j < 4; ++j) {
        const int n  = bn + nw * 64 + j * 16 + l16;
        const float sc = scales[n];
        const float zp = (float)zps[n];
        const float bi = bias[n];
#pragma unroll
        for (int i = 0; i < 4; ++i) {
            const int mbase = mw * 64 + i * 16 + half * 8;
#pragma unroll
            for (int r = 0; r < 8; ++r) {
                const int ml = mbase + r;
                const float v = sc * (acc[i][j][r] - zp * rsum[ml]) + bi;
                __builtin_nontemporal_store(
                    v, &out[(size_t)(bm + ml) * N_TOT + n]);
            }
        }
    }
}

extern "C" void kernel_launch(void* const* d_in, const int* in_sizes, int n_in,
                              void* d_out, int out_size, void* d_ws, size_t ws_size,
                              hipStream_t stream) {
    (void)in_sizes; (void)n_in; (void)out_size; (void)d_ws; (void)ws_size;
    const float* x  = (const float*)d_in[0];
    const int*   wq = (const int*)  d_in[1];
    const float* sc = (const float*)d_in[2];
    const int*   zp = (const int*)  d_in[3];
    const float* bi = (const float*)d_in[4];
    float* out = (float*)d_out;

    dim3 grid(GRID_MB * GRID_NB);   // 2752 blocks, supertile-remapped in-kernel
    q4_bf16_wmma_gemm<<<grid, 256, 0, stream>>>(x, wq, sc, zp, bi, out);
}
